// LLMCC_74414603370526
// MI455X (gfx1250) — matrix operand order repaired and checked by reading, not compiled, not run on hardware
//
#include <hip/hip_runtime.h>
#include <hip/hip_bf16.h>
#include <math.h>

typedef __attribute__((ext_vector_type(16))) _Float16 v16h;
typedef __attribute__((ext_vector_type(8)))  float    v8f;
typedef __attribute__((ext_vector_type(4)))  int      v4i_;

#define B_  16
#define S_  512
#define H_  768
#define NH_ 8
#define HD_ 96
#define NT_ (B_ * S_)   // 8192 tokens

static __device__ __forceinline__ v8f wmma_f16(v16h a, v16h b, v8f c) {
  return __builtin_amdgcn_wmma_f32_16x16x32_f16(false, a, false, b, (short)0, c, false, false);
}

union V16U { uint4 u[2]; v16h v; };

// Async global->LDS copy of 16 bytes (per-lane) with safe fallback.
// Probe-revealed signature: (int4 AS1*, int4 AS3*, imm offset, imm cpol).
static __device__ __forceinline__ void copy16_g2lds(const _Float16* gp, _Float16* lp) {
#if __has_builtin(__builtin_amdgcn_global_load_async_to_lds_b128)
  v4i_* gg = (v4i_*)const_cast<_Float16*>(gp);
  v4i_* ll = (v4i_*)lp;
  __builtin_amdgcn_global_load_async_to_lds_b128(
      (__attribute__((address_space(1))) v4i_*)gg,
      (__attribute__((address_space(3))) v4i_*)ll, 0, 0);
#else
  *(uint4*)lp = *(const uint4*)gp;
#endif
}

#if __has_builtin(__builtin_amdgcn_global_load_async_to_lds_b128)
#if __has_builtin(__builtin_amdgcn_s_wait_asynccnt)
#define WAIT_ASYNC(n) __builtin_amdgcn_s_wait_asynccnt(n)
#else
#define WAIT_ASYNC(n) asm volatile("s_wait_asynccnt " #n ::: "memory")
#endif
#else
#define WAIT_ASYNC(n) ((void)0)
#endif

// ---------------------------------------------------------------------------
// fp32 -> fp16 bulk convert (8 elems/thread, vectorized both directions)
// ---------------------------------------------------------------------------
__global__ __launch_bounds__(256)
void cvt_f16_kernel(const float* __restrict__ src, _Float16* __restrict__ dst, int n8)
{
  int i = blockIdx.x * 256 + threadIdx.x;
  if (i >= n8) return;
  const float4* s = (const float4*)src;
  float4 a = s[2 * i], b = s[2 * i + 1];
  union { uint4 u4; _Float16 h[8]; } o;
  o.h[0] = (_Float16)a.x; o.h[1] = (_Float16)a.y; o.h[2] = (_Float16)a.z; o.h[3] = (_Float16)a.w;
  o.h[4] = (_Float16)b.x; o.h[5] = (_Float16)b.y; o.h[6] = (_Float16)b.z; o.h[7] = (_Float16)b.w;
  ((uint4*)dst)[i] = o.u4;
}

// ---------------------------------------------------------------------------
// Block reduction helper (wave32-aware)
// ---------------------------------------------------------------------------
static __device__ __forceinline__ float block_reduce_sum(float v, float* sh) {
  __syncthreads();
  #pragma unroll
  for (int o = 16; o > 0; o >>= 1) v += __shfl_xor(v, o, 32);
  int lane = threadIdx.x & 31, w = threadIdx.x >> 5;
  if (lane == 0) sh[w] = v;
  __syncthreads();
  int nw = blockDim.x >> 5;
  v = (threadIdx.x < (unsigned)nw) ? sh[threadIdx.x] : 0.0f;
  if (w == 0) {
    #pragma unroll
    for (int o = 4; o > 0; o >>= 1) v += __shfl_xor(v, o, 32);
    if (lane == 0) sh[0] = v;
  }
  __syncthreads();
  return sh[0];
}

// ---------------------------------------------------------------------------
// WMMA GEMM, f16 operands / f32 accumulate:
//   out = A[M,K] @ W[N,K]^T + bias[N] (+ residual[M,N])
// Block: 256 threads (8 waves) -> 64x128 tile; wave -> 16 rows x 64 cols
// (4 accumulators). Double-buffered async(b128) global->LDS staging with
// partial s_wait_asynccnt to overlap DMA with WMMA.
// M multiple of 64, N multiple of 128, K multiple of 32.
// ---------------------------------------------------------------------------
__global__ __launch_bounds__(256)
void wmma_gemm_kernel(const _Float16* __restrict__ A, const _Float16* __restrict__ W,
                      const float* __restrict__ bias, const float* __restrict__ residual,
                      float* __restrict__ outF, _Float16* __restrict__ outH,
                      int M, int N, int K)
{
  __shared__ __attribute__((aligned(16))) _Float16 At[2][64][40];   // rows x k (padded)
  __shared__ __attribute__((aligned(16))) _Float16 Wt[2][128][40];  // cols x k (padded)

  const int tid  = threadIdx.x;
  const int lane = tid & 31;
  const int wave = tid >> 5;
  const int wm   = wave & 3;   // 4 row groups of 16
  const int wn   = wave >> 2;  // 2 col groups of 64
  const int rowBase = blockIdx.y * 64;
  const int colBase = blockIdx.x * 128;
  const int half = lane >> 4;
  const int l15  = lane & 15;

  // Staging slots: A: row = tid>>2 (0..63), kseg=(tid&3)*8 (one 16B copy)
  //                W: row = tid>>1 (0..127), kseg=(tid&1)*16 (two 16B copies)
  const int alr = tid >> 2, alk = (tid & 3) * 8;
  const int wlr = tid >> 1, wlk = (tid & 1) * 16;
  const _Float16* Ag = A + (size_t)(rowBase + alr) * K + alk;
  const _Float16* Wg = W + (size_t)(colBase + wlr) * K + wlk;

  v8f acc[4];
  #pragma unroll
  for (int t = 0; t < 4; ++t) acc[t] = (v8f){};

  const int nk = K >> 5;

  // Prologue: stage tile 0 into buffer 0 (3 async ops per wave)
  copy16_g2lds(Ag, &At[0][alr][alk]);
  copy16_g2lds(Wg, &Wt[0][wlr][wlk]);
  copy16_g2lds(Wg + 8, &Wt[0][wlr][wlk + 8]);

  for (int i = 0; i < nk; ++i) {
    const int cur = i & 1;
    if (i + 1 < nk) {
      // Issue next tile into the other buffer, then drain only the older tile.
      const int k = (i + 1) << 5;
      copy16_g2lds(Ag + k, &At[cur ^ 1][alr][alk]);
      copy16_g2lds(Wg + k, &Wt[cur ^ 1][wlr][wlk]);
      copy16_g2lds(Wg + k + 8, &Wt[cur ^ 1][wlr][wlk + 8]);
      WAIT_ASYNC(3);
    } else {
      WAIT_ASYNC(0);
    }
    __syncthreads();

    // A fragment: lane -> M = wm*16 + l15; elems 0..7 -> K=half*8.., 8..15 -> +16
    const int m  = wm * 16 + l15;
    const int kb = half * 8;
    V16U a;
    a.u[0] = *(const uint4*)&At[cur][m][kb];
    a.u[1] = *(const uint4*)&At[cur][m][kb + 16];

    // B fragments: lane -> N col; elems 0..7 -> K=half*8.., 8..15 -> +16
    const int kcol = half * 8;
    #pragma unroll
    for (int t = 0; t < 4; ++t) {
      const int n0 = wn * 64 + t * 16 + l15;
      V16U bf;
      bf.u[0] = *(const uint4*)&Wt[cur][n0][kcol];
      bf.u[1] = *(const uint4*)&Wt[cur][n0][kcol + 16];
      acc[t] = wmma_f16(a.v, bf.v, acc[t]);
    }
    __syncthreads();  // all waves done reading buf cur before it is re-staged
  }

  // Epilogue: element r -> row = r + 8*half, col = l15
  const int rw = rowBase + wm * 16 + 8 * half;
  #pragma unroll
  for (int t = 0; t < 4; ++t) {
    const int c = colBase + wn * 64 + t * 16 + l15;
    const float bz = bias[c];
    #pragma unroll
    for (int r = 0; r < 8; ++r) {
      size_t o = (size_t)(rw + r) * N + c;
      float v = acc[t][r] + bz;
      if (residual) v += residual[o];
      if (outF) outF[o] = v;
      if (outH) outH[o] = (_Float16)v;
    }
  }
}

// ---------------------------------------------------------------------------
// Fused attention over f16 QKV: one wave per 16 query rows of one (b,h).
// QK^T (WMMA) -> masked/scaled scores in LDS -> in-wave softmax -> probs@V.
// Block = 128 threads (4 waves) -> 128 KB LDS.
// ---------------------------------------------------------------------------
__global__ __launch_bounds__(128)
void attention_kernel(const _Float16* __restrict__ qkv, const int* __restrict__ amask,
                      _Float16* __restrict__ aoh)
{
  __shared__ float sc[4][16][S_];

  const int lane = threadIdx.x & 31;
  const int wave = threadIdx.x >> 5;
  const int b = blockIdx.z;
  const int h = blockIdx.y;
  const int q0 = (blockIdx.x * 4 + wave) * 16;
  const int half = lane >> 4;
  const int mrow = lane & 15;

  const _Float16* base = qkv + (size_t)b * S_ * (3 * H_);

  // Q fragments (16 x 96 -> 3 k-steps) via vector loads
  v16h aq[3];
  #pragma unroll
  for (int ks = 0; ks < 3; ++ks) {
    const _Float16* qr = base + (size_t)(q0 + mrow) * (3 * H_) + h * HD_ + ks * 32 + half * 8;
    V16U t;
    t.u[0] = *(const uint4*)qr;
    t.u[1] = *(const uint4*)(qr + 16);
    aq[ks] = t.v;
  }

  const float scale = 0.1020620726f;  // 1/sqrt(96)

  // Pass 1: score tiles
  for (int n0 = 0; n0 < S_; n0 += 16) {
    const int key = n0 + mrow;
    const _Float16* kr = base + (size_t)key * (3 * H_) + H_ + h * HD_ + half * 8;
    v8f acc = {};
    #pragma unroll
    for (int ks = 0; ks < 3; ++ks) {
      V16U bk;
      bk.u[0] = *(const uint4*)(kr + ks * 32);
      bk.u[1] = *(const uint4*)(kr + ks * 32 + 16);
      acc = wmma_f16(aq[ks], bk.v, acc);
    }
    const bool pad = (amask[b * S_ + key] == 0);
    #pragma unroll
    for (int r = 0; r < 8; ++r) {
      float v = acc[r] * scale;
      if (pad) v = -1e9f;
      sc[wave][r + 8 * half][key] = v;
    }
  }

  // In-wave softmax: lane handles row = mrow, key half [half*256, +256)
  {
    float* row = sc[wave][mrow];
    const int j0 = half * 256;
    float mx = -1e30f;
    for (int j = 0; j < 256; ++j) mx = fmaxf(mx, row[j0 + j]);
    mx = fmaxf(mx, __shfl_xor(mx, 16, 32));
    float sum = 0.0f;
    for (int j = 0; j < 256; ++j) {
      float e = __expf(row[j0 + j] - mx);
      row[j0 + j] = e;
      sum += e;
    }
    sum += __shfl_xor(sum, 16, 32);
    float inv = 1.0f / sum;
    for (int j = 0; j < 256; ++j) row[j0 + j] *= inv;
  }

  // Pass 2: out = probs @ V (16x512 @ 512x96)
  v8f oacc[6];
  #pragma unroll
  for (int t = 0; t < 6; ++t) oacc[t] = (v8f){};

  for (int k0 = 0; k0 < S_; k0 += 32) {
    v16h ap;
    const int kb = k0 + half * 16;
    #pragma unroll
    for (int e = 0; e < 8; ++e) {
      ap[e]     = (_Float16)sc[wave][mrow][kb + e];
      ap[e + 8] = (_Float16)sc[wave][mrow][kb + 8 + e];
    }
    const int keyb = k0 + half * 16;
    #pragma unroll
    for (int nt = 0; nt < 6; ++nt) {
      v16h bv;
      const _Float16* vb = base + (size_t)keyb * (3 * H_) + 2 * H_ + h * HD_ + nt * 16 + mrow;
      #pragma unroll
      for (int e = 0; e < 16; ++e) bv[e] = vb[(size_t)e * (3 * H_)];
      oacc[nt] = wmma_f16(ap, bv, oacc[nt]);
    }
  }

  #pragma unroll
  for (int nt = 0; nt < 6; ++nt) {
    #pragma unroll
    for (int r = 0; r < 8; ++r) {
      aoh[(size_t)(b * S_ + q0 + r + 8 * half) * H_ + h * HD_ + nt * 16 + mrow] =
          (_Float16)oacc[nt][r];
    }
  }
}

// ---------------------------------------------------------------------------
// LayerNorm (+ optional exact GELU). One block (256 thr) per token.
// Reads f32 GEMM output; writes f32 (outF) and/or f16 (outH). in==outF OK.
// ---------------------------------------------------------------------------
__global__ __launch_bounds__(256)
void ln_act_kernel(const float* __restrict__ in, float* __restrict__ outF,
                   _Float16* __restrict__ outH,
                   const float* __restrict__ g, const float* __restrict__ be,
                   int W, int do_gelu)
{
  __shared__ float sh[32];
  const float* row = in + (size_t)blockIdx.x * W;
  const int per = W >> 8;            // W/256 : 4, 2, or 1
  float vals[4];
  float s = 0.0f;
  for (int i = 0; i < per; ++i) { vals[i] = row[threadIdx.x + (i << 8)]; s += vals[i]; }
  const float mean = block_reduce_sum(s, sh) / (float)W;
  float vs = 0.0f;
  for (int i = 0; i < per; ++i) { float d = vals[i] - mean; vs += d * d; }
  const float var = block_reduce_sum(vs, sh) / (float)W;
  const float inv = rsqrtf(var + 1e-5f);
  for (int i = 0; i < per; ++i) {
    int c = threadIdx.x + (i << 8);
    float v = (vals[i] - mean) * inv * g[c] + be[c];
    if (do_gelu) v = 0.5f * v * (1.0f + erff(v * 0.70710678f));
    if (outF) outF[(size_t)blockIdx.x * W + c] = v;
    if (outH) outH[(size_t)blockIdx.x * W + c] = (_Float16)v;
  }
}

// ---------------------------------------------------------------------------
// logits + cross-entropy: one wave per token (8 waves / block).
// ---------------------------------------------------------------------------
__global__ __launch_bounds__(256)
void ce_kernel(const float* __restrict__ feat, const float* __restrict__ emb,
               const float* __restrict__ wc, const float* __restrict__ bc,
               const float* __restrict__ wr, const float* __restrict__ br,
               const int* __restrict__ labels, float* __restrict__ ce_sum)
{
  const int lane = threadIdx.x & 31;
  const int wave = threadIdx.x >> 5;
  const int t = blockIdx.x * 8 + wave;
  const float* f = feat + (size_t)t * 256;
  const float* e = emb + (size_t)t * H_;
  float lg[9];
  #pragma unroll
  for (int l = 0; l < 9; ++l) {
    float s = 0.0f;
    for (int j = lane; j < 256; j += 32) s += f[j] * wc[l * 256 + j];
    float s2 = 0.0f;
    for (int j = lane; j < H_; j += 32) s2 += e[j] * wr[l * H_ + j];
    s += 0.2f * s2;
    #pragma unroll
    for (int o = 16; o > 0; o >>= 1) s += __shfl_xor(s, o, 32);
    lg[l] = s + bc[l] + 0.2f * br[l];
  }
  if (lane == 0) {
    float mx = lg[0];
    #pragma unroll
    for (int l = 1; l < 9; ++l) mx = fmaxf(mx, lg[l]);
    float sum = 0.0f;
    #pragma unroll
    for (int l = 0; l < 9; ++l) sum += __expf(lg[l] - mx);
    const float lse = mx + __logf(sum);
    atomicAdd(ce_sum, lse - lg[labels[t]]);
  }
}

// ---------------------------------------------------------------------------
// Quadruplet loss: one block per batch element.
// ---------------------------------------------------------------------------
__global__ __launch_bounds__(256)
void quad_kernel(const float* __restrict__ emb, const int* __restrict__ ap,
                 const int* __restrict__ pp, const int* __restrict__ n1p,
                 const int* __restrict__ n2p, float* __restrict__ quad_sum)
{
  __shared__ float sh[32];
  const int b = blockIdx.x;
  const float* a  = emb + ((size_t)b * S_ + ap[b])  * H_;
  const float* p  = emb + ((size_t)b * S_ + pp[b])  * H_;
  const float* n1 = emb + ((size_t)b * S_ + n1p[b]) * H_;
  const float* n2 = emb + ((size_t)b * S_ + n2p[b]) * H_;
  float pd = 0.0f, d1 = 0.0f, d2 = 0.0f;
  for (int j = threadIdx.x; j < H_; j += 256) {
    float av = a[j];
    float x0 = av - p[j];  pd += x0 * x0;
    float x1 = av - n1[j]; d1 += x1 * x1;
    float x2 = av - n2[j]; d2 += x2 * x2;
  }
  pd = block_reduce_sum(pd, sh);
  d1 = block_reduce_sum(d1, sh);
  d2 = block_reduce_sum(d2, sh);
  if (threadIdx.x == 0)
    atomicAdd(quad_sum, fmaxf(pd - d1 + 1.0f, 0.0f) + fmaxf(pd - d2 + 0.5f, 0.0f));
}

// ---------------------------------------------------------------------------
// Contextual continuity: one block per (s, b) adjacent pair.
// ---------------------------------------------------------------------------
__global__ __launch_bounds__(256)
void ctx_kernel(const float* __restrict__ emb, const int* __restrict__ labels,
                float* __restrict__ ctx_sum)
{
  __shared__ float sh[32];
  const int s = blockIdx.x;
  const int b = blockIdx.y;
  const float* r0 = emb + ((size_t)b * S_ + s) * H_;
  const float* r1 = r0 + H_;
  float ss = 0.0f;
  for (int j = threadIdx.x; j < H_; j += 256) { float d = r0[j] - r1[j]; ss += d * d; }
  ss = block_reduce_sum(ss, sh);
  if (threadIdx.x == 0) {
    const int l0 = labels[b * S_ + s], l1 = labels[b * S_ + s + 1];
    if (l0 != 0 && l0 == l1) atomicAdd(ctx_sum, sqrtf(ss));
  }
}

__global__ void init_kernel(float* scal) {
  if (threadIdx.x < 4) scal[threadIdx.x] = 0.0f;
}

__global__ void finalize_kernel(const float* __restrict__ scal, float* __restrict__ out) {
  out[0] = scal[0] / (float)NT_
         + 0.2f * (scal[1] / (float)B_)
         + 0.1f * (scal[2] / (float)(B_ * S_));
}

// ---------------------------------------------------------------------------
extern "C" void kernel_launch(void* const* d_in, const int* in_sizes, int n_in,
                              void* d_out, int out_size, void* d_ws, size_t ws_size,
                              hipStream_t stream)
{
  (void)in_sizes; (void)n_in; (void)out_size; (void)ws_size;

  const float* x      = (const float*)d_in[0];
  const int*   amask  = (const int*)d_in[1];
  const int*   labels = (const int*)d_in[2];
  const int*   apos   = (const int*)d_in[3];
  const int*   ppos   = (const int*)d_in[4];
  const int*   n1pos  = (const int*)d_in[5];
  const int*   n2pos  = (const int*)d_in[6];
  const float* w_qkv  = (const float*)d_in[7];
  const float* b_qkv  = (const float*)d_in[8];
  const float* w_o    = (const float*)d_in[9];
  const float* b_o    = (const float*)d_in[10];
  const float* w1     = (const float*)d_in[11];
  const float* b1     = (const float*)d_in[12];
  const float* g1     = (const float*)d_in[13];
  const float* be1    = (const float*)d_in[14];
  const float* w2     = (const float*)d_in[15];
  const float* b2     = (const float*)d_in[16];
  const float* g2     = (const float*)d_in[17];
  const float* be2    = (const float*)d_in[18];
  const float* w3     = (const float*)d_in[19];
  const float* b3     = (const float*)d_in[20];
  const float* g3     = (const float*)d_in[21];
  const float* be3    = (const float*)d_in[22];
  const float* wc     = (const float*)d_in[23];
  const float* bc     = (const float*)d_in[24];
  const float* wr     = (const float*)d_in[25];
  const float* br     = (const float*)d_in[26];

  // Workspace layout (256B-aligned chunks)
  char* p = (char*)d_ws;
  auto alloc = [&](size_t bytes) { void* r = p; p += (bytes + 255) & ~(size_t)255; return r; };
  _Float16* qkvh  = (_Float16*)alloc((size_t)NT_ * 3 * H_ * 2);
  _Float16* aoh   = (_Float16*)alloc((size_t)NT_ * H_ * 2);
  float*    emb   = (float*)   alloc((size_t)NT_ * H_ * 4);
  _Float16* embh  = (_Float16*)alloc((size_t)NT_ * H_ * 2);
  float*    y1    = (float*)   alloc((size_t)NT_ * 1024 * 4);
  _Float16* h1h   = (_Float16*)alloc((size_t)NT_ * 1024 * 2);
  float*    y2    = (float*)   alloc((size_t)NT_ * 512 * 4);
  _Float16* h2h   = (_Float16*)alloc((size_t)NT_ * 512 * 2);
  float*    y3    = (float*)   alloc((size_t)NT_ * 256 * 4);
  _Float16* xh    = (_Float16*)alloc((size_t)NT_ * H_ * 2);
  _Float16* wqkvh = (_Float16*)alloc((size_t)3 * H_ * H_ * 2);
  _Float16* woh   = (_Float16*)alloc((size_t)H_ * H_ * 2);
  _Float16* w1h   = (_Float16*)alloc((size_t)1024 * H_ * 2);
  _Float16* w2h   = (_Float16*)alloc((size_t)512 * 1024 * 2);
  _Float16* w3h   = (_Float16*)alloc((size_t)256 * 512 * 2);
  float*    scal  = (float*)   alloc(4 * 4);

  init_kernel<<<1, 32, 0, stream>>>(scal);

  // f32 -> f16 conversions (activations input + all GEMM weights)
  auto cvt = [&](const float* s, _Float16* d, size_t n) {
    int n8 = (int)(n / 8);
    cvt_f16_kernel<<<(n8 + 255) / 256, 256, 0, stream>>>(s, d, n8);
  };
  cvt(x,     xh,    (size_t)NT_ * H_);
  cvt(w_qkv, wqkvh, (size_t)3 * H_ * H_);
  cvt(w_o,   woh,   (size_t)H_ * H_);
  cvt(w1,    w1h,   (size_t)1024 * H_);
  cvt(w2,    w2h,   (size_t)512 * 1024);
  cvt(w3,    w3h,   (size_t)256 * 512);

  // QKV projection -> f16 qkv
  wmma_gemm_kernel<<<dim3(2304 / 128, NT_ / 64), 256, 0, stream>>>(
      xh, wqkvh, b_qkv, nullptr, nullptr, qkvh, NT_, 3 * H_, H_);

  // Fused attention -> f16 ao
  attention_kernel<<<dim3(S_ / 64, NH_, B_), 128, 0, stream>>>(qkvh, amask, aoh);

  // Output projection with residual: emb = x + ao @ w_o^T + b_o (f32 + f16)
  wmma_gemm_kernel<<<dim3(H_ / 128, NT_ / 64), 256, 0, stream>>>(
      aoh, woh, b_o, x, emb, embh, NT_, H_, H_);

  // Feature extractor
  wmma_gemm_kernel<<<dim3(1024 / 128, NT_ / 64), 256, 0, stream>>>(
      embh, w1h, b1, nullptr, y1, nullptr, NT_, 1024, H_);
  ln_act_kernel<<<NT_, 256, 0, stream>>>(y1, nullptr, h1h, g1, be1, 1024, 1);

  wmma_gemm_kernel<<<dim3(512 / 128, NT_ / 64), 256, 0, stream>>>(
      h1h, w2h, b2, nullptr, y2, nullptr, NT_, 512, 1024);
  ln_act_kernel<<<NT_, 256, 0, stream>>>(y2, nullptr, h2h, g2, be2, 512, 1);

  wmma_gemm_kernel<<<dim3(256 / 128, NT_ / 64), 256, 0, stream>>>(
      h2h, w3h, b3, nullptr, y3, nullptr, NT_, 256, 512);
  ln_act_kernel<<<NT_, 256, 0, stream>>>(y3, y3, nullptr, g3, be3, 256, 0);  // feat (f32, in-place)

  // Losses
  ce_kernel<<<NT_ / 8, 256, 0, stream>>>(y3, emb, wc, bc, wr, br, labels, &scal[0]);
  quad_kernel<<<B_, 256, 0, stream>>>(emb, apos, ppos, n1pos, n2pos, &scal[1]);
  ctx_kernel<<<dim3(S_ - 1, B_), 256, 0, stream>>>(emb, labels, &scal[2]);

  finalize_kernel<<<1, 1, 0, stream>>>(scal, (float*)d_out);
}
